// BoxMultiHeadedAttention_19421842112678
// MI455X (gfx1250) — matrix operand order repaired and checked by reading, not compiled, not run on hardware
//
#include <hip/hip_runtime.h>

// ---------------------------------------------------------------------------
// BoxMultiHeadedAttention for MI455X (gfx1250, wave32, WMMA + async/TDM)
// B=8 N=512 D=512 H=8 DK=64 DIM_G=64
// ---------------------------------------------------------------------------

typedef _Float16 h8  __attribute__((ext_vector_type(8)));
typedef _Float16 h16 __attribute__((ext_vector_type(16)));
typedef float    f8  __attribute__((ext_vector_type(8)));
typedef unsigned int u32x4 __attribute__((ext_vector_type(4)));
typedef int      i32x4 __attribute__((ext_vector_type(4)));
typedef int      i32x8 __attribute__((ext_vector_type(8)));

union Frag { h16 v; h8 h[2]; };

static __device__ __forceinline__ f8 wmma_f16(h16 a, h16 b, f8 c) {
  return __builtin_amdgcn_wmma_f32_16x16x32_f16(
      false, a, false, b, (short)0, c, false, false);
}

#if __has_builtin(__builtin_amdgcn_s_wait_asynccnt)
#define S_WAIT_ASYNCCNT(n) __builtin_amdgcn_s_wait_asynccnt(n)
#else
#define S_WAIT_ASYNCCNT(n) asm volatile("s_wait_asynccnt %0" :: "i"(n) : "memory")
#endif

// one 16B global->LDS async copy (GVS mode: SGPR64 base + VGPR32 byte offset)
static __device__ __forceinline__ void async_cp16(unsigned ldsAddr, unsigned gOff,
                                                  const void* base) {
  asm volatile("global_load_async_to_lds_b128 %0, %1, %2"
               :: "v"(ldsAddr), "v"(gOff), "s"(base) : "memory");
}

// truncate generic LDS pointer -> 32-bit LDS byte address (aperture low bits)
static __device__ __forceinline__ unsigned lds_addr(const void* p) {
  return (unsigned)(unsigned long long)p;
}

// ---------------------------------------------------------------------------
// fp32 -> fp16 convert / weight transpose-convert
// ---------------------------------------------------------------------------
__global__ __launch_bounds__(256) void k_f32_to_f16(const float* __restrict__ in,
                                                    _Float16* __restrict__ out, int n) {
  int i = blockIdx.x * 256 + threadIdx.x;
  if (i < n) out[i] = (_Float16)in[i];
}

__global__ __launch_bounds__(256) void k_transpose_f16(const float* __restrict__ in,
                                                       _Float16* __restrict__ out) {
  int i = blockIdx.x * 256 + threadIdx.x;   // 262144 threads
  int k = i >> 9, n = i & 511;
  out[n * 512 + k] = (_Float16)in[i];
}

// ---------------------------------------------------------------------------
// WMMA GEMM: Y[4096x512] = X @ W + bias, async double-buffered LDS pipeline.
//   mode 0: f16 head-major  (Q,K)   mode 1: f16 transposed (V^T)
//   mode 2: f32 row-major   (final)
// Block 256 thr (8 waves); tile 128x64, BK=32; per tile each thread issues
// exactly 3 async 16B copies -> uniform s_wait_asynccnt 3 pipelining.
// ---------------------------------------------------------------------------
__global__ __launch_bounds__(256) void k_gemm(const _Float16* __restrict__ X,
                                              const _Float16* __restrict__ Wt,
                                              const float* __restrict__ bias,
                                              void* __restrict__ outp, int mode) {
  extern __shared__ char smem[];
  _Float16* buf = (_Float16*)smem;            // 2 x (As[128][40] + Bs[64][40])
  const int BUFH = 128 * 40 + 64 * 40;        // 7680 halves per buffer

  const int tid  = threadIdx.x;
  const int wave = tid >> 5;
  const int lane = tid & 31;
  const int rowBase = blockIdx.x * 128;
  const int colBase = blockIdx.y * 64;

  // staging assignments (per thread, per tile): A = 2x16B, B = 1x16B
  const int ar = tid >> 1, ac = (tid & 1) * 16;        // A: 128 rows x 32 halves
  const int bn = tid >> 2, bc = (tid & 3) * 8;         // B: 64 rows x 32 halves
  const unsigned bufA0 = lds_addr(buf);
  const unsigned ldsAoff = (unsigned)((ar * 40 + ac) * 2);
  const unsigned ldsBoff = (unsigned)((128 * 40 + bn * 40 + bc) * 2);

  f8 acc[4] = {};

#define ISSUE_TILE(k0, bsel)                                                    \
  do {                                                                          \
    const unsigned bb_ = bufA0 + (unsigned)(bsel) * (BUFH * 2);                 \
    const unsigned gA_ = (unsigned)((((rowBase + ar) << 9) + (k0) + ac) * 2);   \
    async_cp16(bb_ + ldsAoff,      gA_,       X);                               \
    async_cp16(bb_ + ldsAoff + 16, gA_ + 16,  X);                               \
    const unsigned gB_ = (unsigned)((((colBase + bn) << 9) + (k0) + bc) * 2);   \
    async_cp16(bb_ + ldsBoff,      gB_,       Wt);                              \
  } while (0)

  ISSUE_TILE(0, 0);

  for (int it = 0; it < 16; ++it) {
    if (it < 15) {
      ISSUE_TILE((it + 1) * 32, (it + 1) & 1);
      S_WAIT_ASYNCCNT(3);           // the 3 just issued may be in flight;
    } else {                        // everything older (tile `it`) is done
      S_WAIT_ASYNCCNT(0);
    }
    __syncthreads();

    _Float16* As = buf + (it & 1) * BUFH;
    _Float16* Bs = As + 128 * 40;

    Frag a;
    const int m   = wave * 16 + (lane & 15);
    const int klo = (lane < 16) ? 0 : 8;
    a.h[0] = *(const h8*)&As[m * 40 + klo];
    a.h[1] = *(const h8*)&As[m * 40 + klo + 16];

#pragma unroll
    for (int t = 0; t < 4; ++t) {
      Frag bf;
      const int n  = t * 16 + (lane & 15);
      const int kb = (lane < 16) ? 0 : 8;
      bf.h[0] = *(const h8*)&Bs[n * 40 + kb];
      bf.h[1] = *(const h8*)&Bs[n * 40 + kb + 16];
      acc[t] = wmma_f16(a.v, bf.v, acc[t]);
    }
    __syncthreads();
  }
#undef ISSUE_TILE

#pragma unroll
  for (int t = 0; t < 4; ++t) {
#pragma unroll
    for (int r = 0; r < 8; ++r) {
      const int row = rowBase + wave * 16 + r + ((lane >> 4) << 3);
      const int col = colBase + t * 16 + (lane & 15);
      const float v = acc[t][r] + bias[col];
      if (mode == 2) {
        ((float*)outp)[(size_t)row * 512 + col] = v;
      } else {
        _Float16* o16 = (_Float16*)outp;
        const int b = row >> 9, n = row & 511, h = col >> 6, dk = col & 63;
        if (mode == 0)
          o16[(((size_t)(b * 8 + h) * 512 + n) << 6) + dk] = (_Float16)v;
        else
          o16[(((size_t)(b * 8 + h) * 64 + dk) << 9) + n] = (_Float16)v;
      }
    }
  }
}

// ---------------------------------------------------------------------------
// Geometry bias: bias[b][h][i][j] = log(clip(relu(emb.WG[h]+bG[h]),1e-6))*obj
//                                   + (mask[j]==0 ? -1e9 : 0)
// ---------------------------------------------------------------------------
__global__ __launch_bounds__(256) void k_geo_bias(const float* __restrict__ box,
                                                  const int* __restrict__ mask,
                                                  const unsigned char* __restrict__ notobj,
                                                  const float* __restrict__ WG,
                                                  const float* __restrict__ bG,
                                                  float* __restrict__ biasout) {
  __shared__ float sWG[512];
  __shared__ float sbG[8];
  const int tid = threadIdx.x;
  sWG[tid]       = WG[tid];
  sWG[tid + 256] = WG[tid + 256];
  if (tid < 8) sbG[tid] = bG[tid];
  __syncthreads();

  const int idx = blockIdx.x * 256 + tid;   // B*N*N threads exactly
  const int b = idx >> 18;
  const int rem = idx & 262143;
  const int i = rem >> 9, j = rem & 511;

  const float4 bi = ((const float4*)box)[b * 512 + i];
  const float4 bj = ((const float4*)box)[b * 512 + j];
  const float cxi = (bi.x + bi.z) * 0.5f, cyi = (bi.y + bi.w) * 0.5f;
  const float wi  = bi.z - bi.x + 1.0f,   hi  = bi.w - bi.y + 1.0f;
  const float cxj = (bj.x + bj.z) * 0.5f, cyj = (bj.y + bj.w) * 0.5f;
  const float wj  = bj.z - bj.x + 1.0f,   hj  = bj.w - bj.y + 1.0f;

  float pos[4];
  pos[0] = __logf(fmaxf(fabsf((cxi - cxj) / wi), 1e-3f));
  pos[1] = __logf(fmaxf(fabsf((cyi - cyj) / hi), 1e-3f));
  pos[2] = __logf(wi / wj);
  pos[3] = __logf(hi / hj);

  const float dimmat[8] = {1.0f, 0.42169650342f, 0.17782794100f, 0.07498942093f,
                           0.03162277660f, 0.01333521432f, 0.00562341325f, 0.00237137371f};

  float acc[8];
#pragma unroll
  for (int h = 0; h < 8; ++h) acc[h] = sbG[h];

#pragma unroll
  for (int p = 0; p < 4; ++p) {
#pragma unroll
    for (int t = 0; t < 8; ++t) {
      const float mm = 100.0f * pos[p] * dimmat[t];
      const float sv = __sinf(mm);
      const float cv = __cosf(mm);
      const int g = p * 8 + t;
#pragma unroll
      for (int h = 0; h < 8; ++h)
        acc[h] = fmaf(sWG[h * 64 + g], sv, fmaf(sWG[h * 64 + 32 + g], cv, acc[h]));
    }
  }

  const float mo = (notobj[b * 512 + i] | notobj[b * 512 + j]) ? 0.0f : 1.0f;
  const float sc = (mask[b * 512 + j] == 0) ? -1e9f : 0.0f;
  const size_t base = (size_t)b * 2097152 + (size_t)i * 512 + j;
#pragma unroll
  for (int h = 0; h < 8; ++h) {
    const float g = __logf(fmaxf(acc[h], 1e-6f));
    biasout[base + (size_t)h * 262144] = g * mo + sc;
  }
}

// ---------------------------------------------------------------------------
// Attention: stage K (512x64) and V^T (64x512) per (b,h) in LDS via the
// Tensor Data Mover (descriptor-driven DMA, TENSORcnt) -- TDM padding
// produces the bank-conflict-free row strides (72 / 520 halves) natively.
// 8 waves x 16 rows, online softmax, WMMA for S=QK^T and O=PV.
// ---------------------------------------------------------------------------
__global__ __launch_bounds__(256) void k_attn(const _Float16* __restrict__ qh,
                                              const _Float16* __restrict__ kh,
                                              const _Float16* __restrict__ vT,
                                              const float* __restrict__ gbias,
                                              _Float16* __restrict__ att16) {
  extern __shared__ char smem[];
  _Float16* Ks = (_Float16*)smem;       // [512][72]  (128B row + 16B TDM pad)
  _Float16* Vs = Ks + 512 * 72;         // [64][520]  (1024B row + 16B TDM pad)
  _Float16* Ps = Vs + 64 * 520;         // [8 waves][16][40]

  const int tid  = threadIdx.x;
  const int wave = tid >> 5;
  const int lane = tid & 31;
  const int bh = blockIdx.x, b = bh >> 3, h = bh & 7;
  const int i0 = blockIdx.y * 128;
  const size_t bhN = (size_t)bh * 512;

  const _Float16* kbase = kh + bhN * 64;
  const _Float16* vbase = vT + bhN * 64;

#if __has_builtin(__builtin_amdgcn_tensor_load_to_lds)
  if (wave == 0) {   // one wave issues both DMA descriptors (EXEC ignored)
    const i32x4 zz4 = {0, 0, 0, 0};
    const i32x8 zz8 = {0, 0, 0, 0, 0, 0, 0, 0};
    // ---- K slab: 2D tensor 64 x 512 (row=64 elems), pad 32 DW -> +4 DW ----
    {
      const unsigned long long ga = (unsigned long long)kbase;
      u32x4 g0 = {1u, lds_addr(Ks), (unsigned)ga,
                  (unsigned)((ga >> 32) & 0x1FFFFFFu) | (2u << 30)};
      const int w0 = (1 << 16) | (1 << 20) | (4 << 22) | (3 << 25);
      i32x8 g1 = {w0, 64 << 16, 512 << 16, 64 << 16, 512, 64, 0, 0};
      __builtin_amdgcn_tensor_load_to_lds(g0, g1, zz4, zz4, zz8, 0);
    }
    // ---- V^T slab: 2D tensor 512 x 64 (row=512 elems), pad 256 DW -> +4 DW
    {
      const unsigned long long ga = (unsigned long long)vbase;
      u32x4 g0 = {1u, lds_addr(Vs), (unsigned)ga,
                  (unsigned)((ga >> 32) & 0x1FFFFFFu) | (2u << 30)};
      const int w0 = (1 << 16) | (1 << 20) | (7 << 22) | (3 << 25);
      i32x8 g1 = {w0, 512 << 16, 64 << 16, 512 << 16, 64, 512, 0, 0};
      __builtin_amdgcn_tensor_load_to_lds(g0, g1, zz4, zz4, zz8, 0);
    }
    __builtin_amdgcn_s_wait_tensorcnt(0);
  }
  __syncthreads();
#else
  // fallback: async global->LDS copies (ASYNCcnt)
  const unsigned ksA = lds_addr(Ks), vsA = lds_addr(Vs);
  for (int t = tid; t < 2048; t += 256) {
    const int j = t >> 2, c = (t & 3) * 16;
    async_cp16(ksA + (unsigned)((j * 72 + c) * 2), (unsigned)((j * 64 + c) * 2), kbase);
  }
  for (int t = tid; t < 2048; t += 256) {
    const int dk = t >> 5, c = (t & 31) * 16;
    async_cp16(vsA + (unsigned)((dk * 520 + c) * 2), (unsigned)((dk * 512 + c) * 2), vbase);
  }
  S_WAIT_ASYNCCNT(0);
  __syncthreads();
#endif

  // Q fragments for this wave's 16 rows
  const int m = lane & 15;
  const int iRow = i0 + wave * 16 + m;
  const _Float16* qrow = qh + (bhN + iRow) * 64;
  Frag qf[2];
#pragma unroll
  for (int c = 0; c < 2; ++c) {
    const int klo = c * 32 + ((lane < 16) ? 0 : 8);
    qf[c].h[0] = *(const h8*)(qrow + klo);
    qf[c].h[1] = *(const h8*)(qrow + klo + 16);
  }

  const float* bb = gbias + (size_t)bh * 262144;
  const int iBase = i0 + wave * 16 + ((lane >> 4) << 3);

  f8 o[4] = {};
  float mrow[8], lrow[8];
#pragma unroll
  for (int r = 0; r < 8; ++r) { mrow[r] = -1e30f; lrow[r] = 0.0f; }

  _Float16* pw = Ps + wave * 16 * 40;   // private per-wave strip

  for (int j0 = 0; j0 < 512; j0 += 32) {
    if (j0 + 32 < 512)
      __builtin_prefetch(&bb[(size_t)iBase * 512 + j0 + 32], 0, 1);
    f8 S[2];
#pragma unroll
    for (int s2 = 0; s2 < 2; ++s2) {
      const int jl = j0 + s2 * 16 + (lane & 15);
      const int kb = (lane < 16) ? 0 : 8;
      Frag kf0, kf1;
      kf0.h[0] = *(const h8*)&Ks[jl * 72 + kb];
      kf0.h[1] = *(const h8*)&Ks[jl * 72 + kb + 16];
      kf1.h[0] = *(const h8*)&Ks[jl * 72 + 32 + kb];
      kf1.h[1] = *(const h8*)&Ks[jl * 72 + 32 + kb + 16];
      f8 c = {};
      c = wmma_f16(qf[0].v, kf0.v, c);
      c = wmma_f16(qf[1].v, kf1.v, c);
#pragma unroll
      for (int r = 0; r < 8; ++r)
        c[r] = fmaf(c[r], 0.125f, bb[(size_t)(iBase + r) * 512 + jl]);
      S[s2] = c;
    }

    // online softmax (row spans 16 lanes of a half-wave)
#pragma unroll
    for (int r = 0; r < 8; ++r) {
      float v = fmaxf(S[0][r], S[1][r]);
      v = fmaxf(v, __shfl_xor(v, 1, 32));
      v = fmaxf(v, __shfl_xor(v, 2, 32));
      v = fmaxf(v, __shfl_xor(v, 4, 32));
      v = fmaxf(v, __shfl_xor(v, 8, 32));
      const float nm   = fmaxf(mrow[r], v);
      const float corr = __expf(mrow[r] - nm);
      mrow[r] = nm;
      const float p0 = __expf(S[0][r] - nm);
      const float p1 = __expf(S[1][r] - nm);
      S[0][r] = p0; S[1][r] = p1;
      float rs = p0 + p1;
      rs += __shfl_xor(rs, 1, 32);
      rs += __shfl_xor(rs, 2, 32);
      rs += __shfl_xor(rs, 4, 32);
      rs += __shfl_xor(rs, 8, 32);
      lrow[r] = lrow[r] * corr + rs;
#pragma unroll
      for (int t = 0; t < 4; ++t) o[t][r] *= corr;
    }

    // P: C-layout f32 -> per-wave LDS -> A-layout f16
#pragma unroll
    for (int r = 0; r < 8; ++r) {
      const int M = r + ((lane >> 4) << 3);
      pw[M * 40 + (lane & 15)]      = (_Float16)S[0][r];
      pw[M * 40 + 16 + (lane & 15)] = (_Float16)S[1][r];
    }
    Frag pf;
    {
      const int kl = (lane < 16) ? 0 : 8;
      pf.h[0] = *(const h8*)&pw[m * 40 + kl];
      pf.h[1] = *(const h8*)&pw[m * 40 + kl + 16];
    }

#pragma unroll
    for (int t = 0; t < 4; ++t) {
      Frag vf;
      const int dk  = t * 16 + (lane & 15);
      const int jb2 = j0 + ((lane < 16) ? 0 : 8);
      vf.h[0] = *(const h8*)&Vs[dk * 520 + jb2];
      vf.h[1] = *(const h8*)&Vs[dk * 520 + jb2 + 16];
      o[t] = wmma_f16(pf.v, vf.v, o[t]);
    }
  }

#pragma unroll
  for (int r = 0; r < 8; ++r) {
    const float inv = 1.0f / lrow[r];
    const int ii = iBase + r;
#pragma unroll
    for (int t = 0; t < 4; ++t) {
      const int col = h * 64 + t * 16 + (lane & 15);
      att16[((size_t)(b * 512 + ii)) * 512 + col] = (_Float16)(o[t][r] * inv);
    }
  }
}

// ---------------------------------------------------------------------------
// Host-side orchestration
// ---------------------------------------------------------------------------
extern "C" void kernel_launch(void* const* d_in, const int* in_sizes, int n_in,
                              void* d_out, int out_size, void* d_ws, size_t ws_size,
                              hipStream_t stream) {
  (void)in_sizes; (void)n_in; (void)out_size; (void)ws_size;

  const float* q32 = (const float*)d_in[0];
  const float* k32 = (const float*)d_in[1];
  const float* v32 = (const float*)d_in[2];
  const float* box = (const float*)d_in[3];
  const int*   msk = (const int*)d_in[4];
  const unsigned char* nob = (const unsigned char*)d_in[5];
  const float* Wq = (const float*)d_in[6];
  const float* bq = (const float*)d_in[7];
  const float* Wk = (const float*)d_in[8];
  const float* bk = (const float*)d_in[9];
  const float* Wv = (const float*)d_in[10];
  const float* bv = (const float*)d_in[11];
  const float* Wo = (const float*)d_in[12];
  const float* bo = (const float*)d_in[13];
  const float* WG = (const float*)d_in[14];
  const float* bG = (const float*)d_in[15];
  float* out = (float*)d_out;

  char* ws = (char*)d_ws;
  const size_t MB = 1024 * 1024;
  _Float16* xq16  = (_Float16*)(ws + 0 * MB);
  _Float16* xk16  = (_Float16*)(ws + 4 * MB);
  _Float16* xv16  = (_Float16*)(ws + 8 * MB);
  _Float16* wqT   = (_Float16*)(ws + 12 * MB);
  _Float16* wkT   = (_Float16*)(ws + 12 * MB + 512 * 1024);
  _Float16* wvT   = (_Float16*)(ws + 13 * MB);
  _Float16* woT   = (_Float16*)(ws + 13 * MB + 512 * 1024);
  _Float16* qh    = (_Float16*)(ws + 14 * MB);
  _Float16* kh    = (_Float16*)(ws + 18 * MB);
  _Float16* vTh   = (_Float16*)(ws + 22 * MB);
  _Float16* att16 = (_Float16*)(ws + 26 * MB);
  float*    gbias = (float*)   (ws + 30 * MB);   // 64 MB

  const int nElem = 8 * 512 * 512;

  k_f32_to_f16<<<nElem / 256, 256, 0, stream>>>(q32, xq16, nElem);
  k_f32_to_f16<<<nElem / 256, 256, 0, stream>>>(k32, xk16, nElem);
  k_f32_to_f16<<<nElem / 256, 256, 0, stream>>>(v32, xv16, nElem);
  k_transpose_f16<<<1024, 256, 0, stream>>>(Wq, wqT);
  k_transpose_f16<<<1024, 256, 0, stream>>>(Wk, wkT);
  k_transpose_f16<<<1024, 256, 0, stream>>>(Wv, wvT);
  k_transpose_f16<<<1024, 256, 0, stream>>>(Wo, woT);

  const size_t gemmLds = (size_t)2 * (128 * 40 + 64 * 40) * sizeof(_Float16); // 30 KB
  dim3 ggrid(32, 8);
  k_gemm<<<ggrid, 256, gemmLds, stream>>>(xq16, wqT, bq, qh, 0);
  k_gemm<<<ggrid, 256, gemmLds, stream>>>(xk16, wkT, bk, kh, 0);
  k_gemm<<<ggrid, 256, gemmLds, stream>>>(xv16, wvT, bv, vTh, 1);

  k_geo_bias<<<8192, 256, 0, stream>>>(box, msk, nob, WG, bG, gbias);

  const size_t attnLds = (size_t)(512 * 72 + 64 * 520 + 8 * 16 * 40) * sizeof(_Float16);
  dim3 agrid(64, 4);
  k_attn<<<agrid, 256, attnLds, stream>>>(qh, kh, vTh, gbias, att16);

  k_gemm<<<ggrid, 256, gemmLds, stream>>>(att16, woT, bo, (void*)out, 2);
}
